// DCNv2PackFlowGuided_53102975647994
// MI455X (gfx1250) — matrix operand order repaired and checked by reading, not compiled, not run on hardware
//
#include <hip/hip_runtime.h>
#include <hip/hip_bf16.h>

typedef __attribute__((ext_vector_type(16))) _Float16 v16h;
typedef __attribute__((ext_vector_type(8)))  _Float16 v8h;
typedef __attribute__((ext_vector_type(4)))  _Float16 v4h;
typedef __attribute__((ext_vector_type(8)))  float    v8f;

static constexpr int Bn   = 4;
static constexpr int Hh   = 128;
static constexpr int Wc   = 128;
static constexpr int NPIX = Bn * Hh * Wc;         // 65536
static constexpr int CIN0 = 160;                  // 130 padded to 5x32
static constexpr int COUT3  = 432;
static constexpr int COUT3P = 448;                // padded to 7x64

// tanh without libcall: gfx1250 has no ftanh lowering; exp lowers natively.
__device__ __forceinline__ float dev_tanhf(float x) {
    float xc = __builtin_fminf(__builtin_fmaxf(x, -15.f), 15.f);
    float e  = __builtin_expf(2.f * xc);
    return 1.f - 2.f / (e + 1.f);
}

__device__ __forceinline__ v16h cat8(v8h lo, v8h hi) {
    return __builtin_shufflevector(lo, hi, 0,1,2,3,4,5,6,7,8,9,10,11,12,13,14,15);
}

// CDNA5 async global->LDS DMA (ASYNCcnt path, ISA 08_async_tensor §4).
// GV mode: VDST = LDS byte offset (addr[31:0] of the shared pointer),
// VADDR = 64-bit global address. 16B per enabled lane, no VGPR data.
__device__ __forceinline__ void async_copy_b128_to_lds(void* lds_ptr, const void* gptr) {
    unsigned int loff = (unsigned int)(unsigned long long)lds_ptr;
    asm volatile("global_load_async_to_lds_b128 %0, %1, off"
                 :: "v"(loff), "v"(gptr) : "memory");
}
__device__ __forceinline__ void wait_async_zero() {
    asm volatile("s_wait_asynccnt 0" ::: "memory");
}

// ---------------------------------------------------------------------------
// Packing kernels: NCHW f32 -> NHWC f16 (channels contiguous for GEMM K dim)
// ---------------------------------------------------------------------------
__global__ void pack_concat_kernel(const float* __restrict__ xfw,
                                   const float* __restrict__ xcur,
                                   const float* __restrict__ flow,
                                   _Float16* __restrict__ Xc) {
    int idx = blockIdx.x * 256 + threadIdx.x;          // over NPIX*CIN0
    if (idx >= NPIX * CIN0) return;
    int pix = idx / CIN0;
    int c   = idx - pix * CIN0;
    int b   = pix >> 14;                               // / (H*W)
    int rem = pix & 16383;                             // h*W + w
    float v = 0.f;
    if (c < 64)       v = xfw [(b * 64 + c)        * 16384 + rem];
    else if (c < 128) v = xcur[(b * 64 + (c - 64)) * 16384 + rem];
    else if (c < 130) v = flow[(b * 2  + (c - 128)) * 16384 + rem];
    Xc[idx] = (_Float16)v;
}

__global__ void pack_x_kernel(const float* __restrict__ x, _Float16* __restrict__ xg) {
    int idx = blockIdx.x * 256 + threadIdx.x;          // over NPIX*64
    if (idx >= NPIX * 64) return;
    int pix = idx >> 6;
    int c   = idx & 63;
    int b   = pix >> 14;
    int rem = pix & 16383;
    xg[idx] = (_Float16)x[(b * 64 + c) * 16384 + rem];
}

// [O][Ci][3][3] f32  ->  [tap][Opad][Cipad] f16, zero padded
__global__ void pack_weights_kernel(const float* __restrict__ src, _Float16* __restrict__ dst,
                                    int O, int Op, int Ci, int Cip, int total) {
    int idx = blockIdx.x * 256 + threadIdx.x;
    if (idx >= total) return;
    int t   = idx / (Op * Cip);
    int rem = idx - t * (Op * Cip);
    int o   = rem / Cip;
    int c   = rem - o * Cip;
    float v = 0.f;
    if (o < O && c < Ci) v = src[(o * Ci + c) * 9 + t];
    dst[idx] = (_Float16)v;
}

// ---------------------------------------------------------------------------
// Implicit-GEMM 3x3 conv via v_wmma_f32_16x16x32_f16.
// WG = 256 threads = 8 waves; tile = 64 pixels x 64 out-channels.
// Wave w -> (mi = w>>1 in 0..3, nh = w&1): TWO 16x16 C tiles (cols nh*32 and
// nh*32+16) sharing one A fragment -> 2 WMMA per A-frag load.
// Whole per-tap K (CIN) staged in LDS via async global->LDS DMA:
// 2 barriers per tap, CIN/32*2 WMMAs in between.
// MODE 0: y = lrelu(conv) -> NHWC f16              (conv0/1/2)
// MODE 1: fused offset/mask epilogue -> P_y/P_x/P_m fp32  (conv3)
// ---------------------------------------------------------------------------
template <int CIN, int COUTP, int MODE>
__global__ __launch_bounds__(256) void conv3x3_wmma_kernel(
    const _Float16* __restrict__ X, const _Float16* __restrict__ Wp,
    const float* __restrict__ bias, _Float16* __restrict__ Y,
    float* __restrict__ Py, float* __restrict__ Px, float* __restrict__ Pm,
    const float* __restrict__ flow) {

    __shared__ __align__(16) _Float16 sA[64 * CIN];   // [pixel][k]
    __shared__ __align__(16) _Float16 sB[64 * CIN];   // [outch][k]

    const int t    = threadIdx.x;
    const int lane = t & 31;
    const int wid  = t >> 5;
    const int mi   = wid >> 1;                       // 0..3  (16-pixel row block)
    const int nh   = wid & 1;                        // 0..1  (32-channel col block)

    const int pix0 = blockIdx.x * 64;                // 64 pixels along W (128%64==0)
    const int nblk = blockIdx.y * 64;
    const int b    = pix0 >> 14;
    const int rem  = pix0 & 16383;
    const int y0   = rem >> 7;
    const int x0   = rem & 127;

    const int ap = t >> 2, aq = t & 3;               // A fill: pixel, 8-half seg
    const int bn = t >> 2, bs = t & 3;               // B fill: out-ch, 8-half seg

    v8f acc0 = {}, acc1 = {};

    for (int tap = 0; tap < 9; ++tap) {
        const int ty = tap / 3 - 1, tx = tap % 3 - 1;
        const int yy = y0 + ty;
        const int xx = x0 + ap + tx;
        const bool inb = (yy >= 0) && (yy < Hh) && (xx >= 0) && (xx < Wc);
        const _Float16* arow = X + (size_t)((b * Hh + yy) * Wc + xx) * CIN;
        const _Float16* brow = Wp + ((size_t)tap * COUTP + nblk + bn) * CIN;

        // prefetch next tap's activation row (global_prefetch_b8)
        if (tap < 8) {
            const int nyy = y0 + (tap + 1) / 3 - 1;
            const int nxx = x0 + ap + (tap + 1) % 3 - 1;
            if (nyy >= 0 && nyy < Hh && nxx >= 0 && nxx < Wc)
                __builtin_prefetch(X + (size_t)((b * Hh + nyy) * Wc + nxx) * CIN, 0, 1);
        }

        // ---- stage full per-tap K: A (64 px x CIN) + B (64 n x CIN)
        // In-bounds lanes + weights go through the async LDS DMA engine;
        // halo lanes zero-fill with a plain ds store.
#pragma unroll
        for (int c0 = 0; c0 < CIN; c0 += 32) {
            if (inb) {
                async_copy_b128_to_lds(&sA[ap * CIN + c0 + aq * 8], arow + c0 + aq * 8);
            } else {
                v8h z = {};
                *(v8h*)&sA[ap * CIN + c0 + aq * 8] = z;
            }
            async_copy_b128_to_lds(&sB[bn * CIN + c0 + bs * 8], brow + c0 + bs * 8);
        }
        wait_async_zero();
        __syncthreads();

        // ---- fragments in contiguous k order (k = 16*half + s), same for A and B; 2 WMMAs per K chunk
        const int fr = lane & 15, fhi = lane >> 4;
        const int ar  = (mi * 16 + fr) * CIN;
        const int br0 = (nh * 32 + fr) * CIN;
        const int br1 = (nh * 32 + 16 + fr) * CIN;
#pragma unroll
        for (int c0 = 0; c0 < CIN; c0 += 32) {
            v16h af = cat8(*(const v8h*)&sA[ar + c0 + fhi * 16],
                           *(const v8h*)&sA[ar + c0 + fhi * 16 + 8]);
            v16h bf0 = cat8(*(const v8h*)&sB[br0 + c0 + fhi * 16],
                            *(const v8h*)&sB[br0 + c0 + fhi * 16 + 8]);
            acc0 = __builtin_amdgcn_wmma_f32_16x16x32_f16(
                false, af, false, bf0, (short)0, acc0, false, false);
            v16h bf1 = cat8(*(const v8h*)&sB[br1 + c0 + fhi * 16],
                            *(const v8h*)&sB[br1 + c0 + fhi * 16 + 8]);
            acc1 = __builtin_amdgcn_wmma_f32_16x16x32_f16(
                false, af, false, bf1, (short)0, acc1, false, false);
        }
        __syncthreads();
    }

    // ---- epilogue: C layout -> lane holds col N=lane&15, VGPR r -> row r+8*(lane>>4)
    const int ehi = lane >> 4, col = lane & 15;
    auto epi = [&](const v8f& a, int ch) {
#pragma unroll
        for (int r = 0; r < 8; ++r) {
            const int pixel = pix0 + mi * 16 + r + ehi * 8;
            if constexpr (MODE == 0) {
                float v = a[r] + bias[ch];
                v = v >= 0.f ? v : 0.1f * v;
                Y[(size_t)pixel * 64 + ch] = (_Float16)v;
            } else {
                if (ch < COUT3) {
                    float v = a[r] + bias[ch];
                    const int pb = pixel >> 14, prem = pixel & 16383;
                    const int py = prem >> 7, px = prem & 127;
                    if (ch < 288) {
                        // offset channel: g*(9*2)+k*2+comp ; comp0=dy(+flow[1]+ky+h)
                        const int g = ch / 18, rr = ch % 18, k = rr >> 1, comp = ch & 1;
                        const float fl = flow[((pb * 2 + (1 - comp)) * Hh + py) * Wc + px];
                        const float v2 = 10.f * dev_tanhf(v) + fl;
                        const int o = pixel * 144 + g * 9 + k;
                        if (comp == 0) Py[o] = v2 + (float)(k / 3 - 1) + (float)py;
                        else           Px[o] = v2 + (float)(k % 3 - 1) + (float)px;
                    } else {
                        const int mc = ch - 288, g = mc / 9, k = mc - g * 9;
                        Pm[pixel * 144 + g * 9 + k] = 1.f / (1.f + __builtin_expf(-v));
                    }
                }
            }
        }
    };
    epi(acc0, nblk + nh * 32 + col);
    epi(acc1, nblk + nh * 32 + 16 + col);
}

// ---------------------------------------------------------------------------
// Deformable conv: prologue bilinear-samples x (NHWC f16, 4-channel deform
// groups share sampling coords) into the LDS A tile, weights via async LDS
// DMA, then same WMMA mainloop over K = 9 taps * 64 channels.
// Output fp32 NCHW into d_out.
// ---------------------------------------------------------------------------
__global__ __launch_bounds__(256) void dcn_wmma_kernel(
    const _Float16* __restrict__ xg,
    const float* __restrict__ Py, const float* __restrict__ Px,
    const float* __restrict__ Pm,
    const _Float16* __restrict__ Wp, const float* __restrict__ bias,
    float* __restrict__ out) {

    __shared__ __align__(16) _Float16 sA[64 * 64];
    __shared__ __align__(16) _Float16 sB[64 * 64];

    const int t    = threadIdx.x;
    const int lane = t & 31;
    const int wid  = t >> 5;
    const int mi   = wid >> 1;
    const int nh   = wid & 1;

    const int pix0 = blockIdx.x * 64;
    const int b    = pix0 >> 14;

    const int bn = t >> 2, bs = t & 3;

    v8f acc0 = {}, acc1 = {};

    for (int tap = 0; tap < 9; ++tap) {
        // ---- B tile (64 n x 64 k) through the async DMA engine first
        const _Float16* brow = Wp + ((size_t)tap * 64 + bn) * 64;
#pragma unroll
        for (int c0 = 0; c0 < 64; c0 += 32)
            async_copy_b128_to_lds(&sB[bn * 64 + c0 + bs * 8], brow + c0 + bs * 8);

        // ---- build sampled A tile: 64 px x 16 deform groups x 4 channels
#pragma unroll
        for (int rep = 0; rep < 4; ++rep) {
            const int task  = rep * 256 + t;       // 0..1023
            const int px_i  = task >> 4;           // pixel 0..63
            const int g     = task & 15;           // deform group
            const int cb    = g * 4;               // channel base
            const int pixel = pix0 + px_i;
            const int idx   = pixel * 144 + g * 9 + tap;
            const float py = Py[idx], px = Px[idx], m = Pm[idx];
            const float y0f = __builtin_floorf(py), x0f = __builtin_floorf(px);
            const float ly = py - y0f, lx = px - x0f;
            const int iy0 = (int)y0f, ix0 = (int)x0f;
            float s0 = 0.f, s1 = 0.f, s2 = 0.f, s3 = 0.f;
            auto corner = [&](int yy, int xx, float wgt) {
                if (yy >= 0 && yy < Hh && xx >= 0 && xx < Wc) {
                    v4h v = *(const v4h*)&xg[(((size_t)(b * Hh + yy)) * Wc + xx) * 64 + cb];
                    s0 += wgt * (float)v[0];
                    s1 += wgt * (float)v[1];
                    s2 += wgt * (float)v[2];
                    s3 += wgt * (float)v[3];
                }
            };
            corner(iy0,     ix0,     (1.f - ly) * (1.f - lx));
            corner(iy0,     ix0 + 1, (1.f - ly) * lx);
            corner(iy0 + 1, ix0,     ly * (1.f - lx));
            corner(iy0 + 1, ix0 + 1, ly * lx);
            v4h sv = { (_Float16)(s0 * m), (_Float16)(s1 * m),
                       (_Float16)(s2 * m), (_Float16)(s3 * m) };
            *(v4h*)&sA[px_i * 64 + cb] = sv;
        }
        wait_async_zero();
        __syncthreads();

        const int fr = lane & 15, fhi = lane >> 4;
        const int ar  = (mi * 16 + fr) * 64;
        const int br0 = (nh * 32 + fr) * 64;
        const int br1 = (nh * 32 + 16 + fr) * 64;
#pragma unroll
        for (int c0 = 0; c0 < 64; c0 += 32) {
            v16h af = cat8(*(const v8h*)&sA[ar + c0 + fhi * 16],
                           *(const v8h*)&sA[ar + c0 + fhi * 16 + 8]);
            v16h bf0 = cat8(*(const v8h*)&sB[br0 + c0 + fhi * 16],
                            *(const v8h*)&sB[br0 + c0 + fhi * 16 + 8]);
            acc0 = __builtin_amdgcn_wmma_f32_16x16x32_f16(
                false, af, false, bf0, (short)0, acc0, false, false);
            v16h bf1 = cat8(*(const v8h*)&sB[br1 + c0 + fhi * 16],
                            *(const v8h*)&sB[br1 + c0 + fhi * 16 + 8]);
            acc1 = __builtin_amdgcn_wmma_f32_16x16x32_f16(
                false, af, false, bf1, (short)0, acc1, false, false);
        }
        __syncthreads();
    }

    const int ehi = lane >> 4, col = lane & 15;
    auto epi = [&](const v8f& a, int ch) {
#pragma unroll
        for (int r = 0; r < 8; ++r) {
            const int pixel = pix0 + mi * 16 + r + ehi * 8;
            const int pb = pixel >> 14, prem = pixel & 16383;
            const int oy = prem >> 7, ox = prem & 127;
            out[(((size_t)pb * 64 + ch) * Hh + oy) * Wc + ox] = a[r] + bias[ch];
        }
    };
    epi(acc0, nh * 32 + col);
    epi(acc1, nh * 32 + 16 + col);
}

// ---------------------------------------------------------------------------
extern "C" void kernel_launch(void* const* d_in, const int* in_sizes, int n_in,
                              void* d_out, int out_size, void* d_ws, size_t ws_size,
                              hipStream_t stream) {
    (void)in_sizes; (void)n_in; (void)out_size; (void)ws_size;
    const float* x    = (const float*)d_in[0];
    const float* xfw  = (const float*)d_in[1];
    const float* xcur = (const float*)d_in[2];
    const float* flow = (const float*)d_in[3];
    const float* w0   = (const float*)d_in[4];
    const float* b0   = (const float*)d_in[5];
    const float* w1   = (const float*)d_in[6];
    const float* b1   = (const float*)d_in[7];
    const float* w2   = (const float*)d_in[8];
    const float* b2   = (const float*)d_in[9];
    const float* w3   = (const float*)d_in[10];
    const float* b3   = (const float*)d_in[11];
    const float* wd   = (const float*)d_in[12];
    const float* bd   = (const float*)d_in[13];

    char* ws = (char*)d_ws;
    size_t off = 0;
    auto alloc = [&](size_t bytes) -> void* {
        void* p = ws + off;
        off = (off + bytes + 255) & ~(size_t)255;
        return p;
    };
    _Float16* Xc  = (_Float16*)alloc((size_t)NPIX * CIN0 * 2);
    _Float16* xg  = (_Float16*)alloc((size_t)NPIX * 64 * 2);
    _Float16* h1  = (_Float16*)alloc((size_t)NPIX * 64 * 2);
    _Float16* h2  = (_Float16*)alloc((size_t)NPIX * 64 * 2);
    float*    Py  = (float*)alloc((size_t)NPIX * 144 * 4);
    float*    Px  = (float*)alloc((size_t)NPIX * 144 * 4);
    float*    Pm  = (float*)alloc((size_t)NPIX * 144 * 4);
    _Float16* w0p = (_Float16*)alloc((size_t)9 * 64 * CIN0 * 2);
    _Float16* w1p = (_Float16*)alloc((size_t)9 * 64 * 64 * 2);
    _Float16* w2p = (_Float16*)alloc((size_t)9 * 64 * 64 * 2);
    _Float16* w3p = (_Float16*)alloc((size_t)9 * COUT3P * 64 * 2);
    _Float16* wdp = (_Float16*)alloc((size_t)9 * 64 * 64 * 2);

    // pack activations + weights
    pack_concat_kernel<<<(NPIX * CIN0) / 256, 256, 0, stream>>>(xfw, xcur, flow, Xc);
    pack_x_kernel<<<(NPIX * 64) / 256, 256, 0, stream>>>(x, xg);
    auto wgrid = [](int total) { return (total + 255) / 256; };
    pack_weights_kernel<<<wgrid(9 * 64 * CIN0), 256, 0, stream>>>(w0, w0p, 64, 64, 130, CIN0, 9 * 64 * CIN0);
    pack_weights_kernel<<<wgrid(9 * 64 * 64), 256, 0, stream>>>(w1, w1p, 64, 64, 64, 64, 9 * 64 * 64);
    pack_weights_kernel<<<wgrid(9 * 64 * 64), 256, 0, stream>>>(w2, w2p, 64, 64, 64, 64, 9 * 64 * 64);
    pack_weights_kernel<<<wgrid(9 * COUT3P * 64), 256, 0, stream>>>(w3, w3p, COUT3, COUT3P, 64, 64, 9 * COUT3P * 64);
    pack_weights_kernel<<<wgrid(9 * 64 * 64), 256, 0, stream>>>(wd, wdp, 64, 64, 64, 64, 9 * 64 * 64);

    const dim3 blk(256);
    const int  mgrid = NPIX / 64;   // 1024 pixel tiles (64 px along W each)

    conv3x3_wmma_kernel<CIN0, 64, 0><<<dim3(mgrid, 1), blk, 0, stream>>>(
        Xc, w0p, b0, h1, nullptr, nullptr, nullptr, nullptr);
    conv3x3_wmma_kernel<64, 64, 0><<<dim3(mgrid, 1), blk, 0, stream>>>(
        h1, w1p, b1, h2, nullptr, nullptr, nullptr, nullptr);
    conv3x3_wmma_kernel<64, 64, 0><<<dim3(mgrid, 1), blk, 0, stream>>>(
        h2, w2p, b2, h1, nullptr, nullptr, nullptr, nullptr);
    conv3x3_wmma_kernel<64, COUT3P, 1><<<dim3(mgrid, COUT3P / 64), blk, 0, stream>>>(
        h1, w3p, b3, nullptr, Py, Px, Pm, flow);
    dcn_wmma_kernel<<<dim3(mgrid, 1), blk, 0, stream>>>(
        xg, Py, Px, Pm, wdp, bd, (float*)d_out);
}
